// GNN_3D_Encoder_43843026157658
// MI455X (gfx1250) — matrix operand order repaired and checked by reading, not compiled, not run on hardware
//
#include <hip/hip_runtime.h>
#include <hip/hip_bf16.h>
#include <math.h>

#define Nn   10000
#define Ee   160000
#define ET   (Ee + Nn)      // 170000 edges incl. self loops
#define Hh   128
#define HC   512            // HEADS * C
#define NG   20
#define LL   3
#define TILES_PER_BLK 5     // 80 edges per block; ET/(16*5) = 2125 exactly

typedef __attribute__((ext_vector_type(16))) _Float16 v16h;
typedef __attribute__((ext_vector_type(8)))  _Float16 v8h;
typedef __attribute__((ext_vector_type(8)))  float    v8f;

// ---------- helpers ----------

__device__ __forceinline__ unsigned fenc(float f) {
    unsigned u = __float_as_uint(f);
    return (u & 0x80000000u) ? ~u : (u | 0x80000000u);
}
__device__ __forceinline__ float fdec(unsigned u) {
    unsigned b = (u & 0x80000000u) ? (u & 0x7FFFFFFFu) : ~u;
    return __uint_as_float(b);
}
__device__ __forceinline__ float red16(float v) {
    v += __shfl_xor(v, 1, 32); v += __shfl_xor(v, 2, 32);
    v += __shfl_xor(v, 4, 32); v += __shfl_xor(v, 8, 32);
    return v;
}
__device__ __forceinline__ float red32(float v) {
    v = red16(v); v += __shfl_xor(v, 16, 32); return v;
}

// A fragment: 16x32 f16 tile of row-major matrix (row stride 128 halves).
// lanes 0-15: rows 0-15, K {k0..k0+15}
// lanes16-31: rows 0-15, K {k0+16..k0+31}
__device__ __forceinline__ v16h load_fragA(const _Float16* M, int row0, int k0, int lane) {
    int grp = lane >> 4, r = lane & 15;
    const _Float16* p = M + (size_t)(row0 + r) * Hh + k0 + grp * 16;
    v8h lo = *(const v8h*)p;
    v8h hi = *(const v8h*)(p + 8);
    v16h a;
#pragma unroll
    for (int i = 0; i < 8; i++) { a[i] = lo[i]; a[i + 8] = hi[i]; }
    return a;
}
// B fragment: 32x16 f16 tile of col-major weight WT[col][k] (col stride 128 halves).
// lane = column; lanes 0-15 hold K k0..k0+15, lanes 16-31 hold K k0+16..k0+31.
__device__ __forceinline__ v16h load_fragB(const _Float16* WT, int n0, int k0, int lane) {
    int grp = lane >> 4, c = lane & 15;
    return *(const v16h*)(WT + (size_t)(n0 + c) * Hh + k0 + grp * 16);
}

// ---------- kernels ----------

// h = emb[clip(z,0,99)], also f16 copy
__global__ void k_embed(const int* z, const float* emb, float* hF32, _Float16* hF16) {
    int i = blockIdx.x * 256 + threadIdx.x;
    if (i >= Nn * Hh) return;
    int n = i >> 7, c = i & 127;
    int zz = min(max(z[n], 0), 99);
    float v = emb[zz * Hh + c];
    hF32[i] = v;
    hF16[i] = (_Float16)v;
}

// per-edge gaussian expansion + ea = ge@Wd + bd ; accumulate self-loop mean numerator
__global__ void k_edge_feat(const float* pos, const int* ei, const float* Wd, const float* bd,
                            _Float16* ea2, float* loopSum, float* cnt) {
    int e = blockIdx.x;
    int t = threadIdx.x;           // 128 threads = channels
    __shared__ float ge[NG];
    __shared__ float sdist;
    int s = ei[e], d = ei[Ee + e];
    if (t == 0) {
        float dx = pos[s * 3 + 0] - pos[d * 3 + 0];
        float dy = pos[s * 3 + 1] - pos[d * 3 + 1];
        float dz = pos[s * 3 + 2] - pos[d * 3 + 2];
        sdist = sqrtf(dx * dx + dy * dy + dz * dz + 1e-12f);
    }
    __syncthreads();
    if (t < NG) {
        float mu = (10.0f / 19.0f) * (float)t;
        float df = sdist - mu;
        ge[t] = expf(-(df * df) / (2.0f * 0.25f + 1e-8f));  // sigma=0.5
    }
    __syncthreads();
    float acc = bd[t];
#pragma unroll
    for (int j = 0; j < NG; j++) acc += ge[j] * Wd[j * Hh + t];
    ea2[(size_t)e * Hh + t] = (_Float16)acc;
    atomicAdd(&loopSum[(size_t)d * Hh + t], acc);
    if (t == 0) atomicAdd(&cnt[d], 1.0f);
}

// self-loop edge features: mean of incoming ea by dst
__global__ void k_loop_div(const float* loopSum, const float* cnt, _Float16* ea2) {
    int i = blockIdx.x * 256 + threadIdx.x;
    if (i >= Nn * Hh) return;
    int n = i >> 7;
    float c = fmaxf(cnt[n], 1.0f);
    ea2[(size_t)(Ee + n) * Hh + (i & 127)] = (_Float16)(loopSum[i] / c);
}

// convert one 128x512 f32 row-major weight to f16 col-major [512][128]
__global__ void k_wt(const float* W, _Float16* WT) {
    int i = blockIdx.x * 256 + threadIdx.x;
    if (i >= HC * Hh) return;
    int col = i >> 7, k = i & 127;
    WT[i] = (_Float16)W[(size_t)k * HC + col];
}

// node projection: out[N,512] = hF16[N,128] @ WT^T + bias  (WMMA f16)
__global__ void k_gemm_proj(const _Float16* Af16, const _Float16* WT, const float* bias,
                            float* out) {
    int row0 = blockIdx.x * 16;
    int wave = threadIdx.x >> 5;
    int lane = threadIdx.x & 31;
    int n0 = (blockIdx.y * 4 + wave) * 16;
    int grp = lane >> 4, r = lane & 15;
    v8f c = {};
#pragma unroll
    for (int kk = 0; kk < 4; kk++) {
        v16h a = load_fragA(Af16, row0, kk * 32, lane);
        v16h b = load_fragB(WT, n0, kk * 32, lane);
        c = __builtin_amdgcn_wmma_f32_16x16x32_f16(false, a, false, b, (short)0, c, false, false);
    }
    int col = n0 + r;
    float bv = bias[col];
#pragma unroll
    for (int m = 0; m < 8; m++)
        out[(size_t)(row0 + m + grp * 8) * HC + col] = c[m] + bv;
}

// fused: e = ea2@We (WMMA, B staged in LDS via async copy), m = leaky(xl[src]+xr[dst]+e),
// logit = m . att[h].  One block = 4 waves (one head each) x TILES_PER_BLK 16-edge tiles.
__global__ void k_edge_logits(const _Float16* ea2, const _Float16* WeT,
                              const float* xl, const float* xr, const float* attL,
                              const int* ei, float* logits, unsigned* amax) {
    extern __shared__ _Float16 sW[];   // 512*128 f16 = 128 KB staged WeT (col-major)
    int tid = threadIdx.x;
    int h = tid >> 5;
    int lane = tid & 31;
    int grp = lane >> 4, rr = lane & 15;

    // ---- stage WeT into LDS with CDNA5 async global->LDS copies (ASYNCcnt) ----
    {
        unsigned ldsBase = (unsigned)(uintptr_t)&sW[0];   // low 32 bits = LDS byte offset
        const char* gsrc = (const char*)WeT;
#pragma unroll 4
        for (int i = 0; i < 64; i++) {                    // 64 * 128 lanes * 16B = 128 KB
            unsigned off = (unsigned)(i * 128 + tid) * 16u;
            unsigned dst = ldsBase + off;
            unsigned long long ga = (unsigned long long)(uintptr_t)(gsrc + off);
            asm volatile("global_load_async_to_lds_b128 %0, %1, off"
                         :: "v"(dst), "v"(ga) : "memory");
        }
        asm volatile("s_wait_asynccnt 0" ::: "memory");
        __syncthreads();
    }

    for (int t = 0; t < TILES_PER_BLK; t++) {
        int e0 = (blockIdx.x * TILES_PER_BLK + t) * 16;

        v16h A[4];
#pragma unroll
        for (int kk = 0; kk < 4; kk++) A[kk] = load_fragA(ea2, e0, kk * 32, lane);

        int sArr[8], dArr[8];
#pragma unroll
        for (int m = 0; m < 8; m++) {
            int e = e0 + m + grp * 8;
            if (e < Ee) { sArr[m] = ei[e]; dArr[m] = ei[Ee + e]; }
            else        { sArr[m] = e - Ee; dArr[m] = e - Ee; }
        }
        float logit[8];
#pragma unroll
        for (int m = 0; m < 8; m++) logit[m] = 0.0f;

#pragma unroll
        for (int ct = 0; ct < 8; ct++) {
            int n0 = h * 128 + ct * 16;
            v8f c = {};
#pragma unroll
            for (int kk = 0; kk < 4; kk++) {
                // B fragment from LDS (ds loads)
                v16h b = *(const v16h*)(&sW[(size_t)(n0 + rr) * Hh + kk * 32 + grp * 16]);
                c = __builtin_amdgcn_wmma_f32_16x16x32_f16(false, A[kk], false, b,
                                                           (short)0, c, false, false);
            }
            int col = n0 + rr;
            float attv = attL[col];   // attL = att[layer] laid out [4][128]
#pragma unroll
            for (int m = 0; m < 8; m++) {
                float val = c[m] + xl[(size_t)sArr[m] * HC + col] + xr[(size_t)dArr[m] * HC + col];
                val = (val > 0.0f) ? val : 0.2f * val;
                logit[m] += val * attv;
            }
        }
        // reduce across the 16 lanes (columns) of each half-wave
#pragma unroll
        for (int m = 0; m < 8; m++) logit[m] = red16(logit[m]);

        if (rr == 0) {
#pragma unroll
            for (int m = 0; m < 8; m++) {
                int e = e0 + m + grp * 8;
                logits[(size_t)e * 4 + h] = logit[m];
                atomicMax(&amax[(size_t)dArr[m] * 4 + h], fenc(logit[m]));
            }
        }
    }
}

// p = exp(logit - amax[dst]); den[dst] += p
__global__ void k_softmax(const float* logits, const unsigned* amax, float* pvals, float* den,
                          const int* ei) {
    int i = blockIdx.x * 256 + threadIdx.x;
    if (i >= ET * 4) return;
    int e = i >> 2, h = i & 3;
    int d = (e < Ee) ? ei[Ee + e] : e - Ee;
    float p = expf(logits[i] - fdec(amax[(size_t)d * 4 + h]));
    pvals[i] = p;
    atomicAdd(&den[(size_t)d * 4 + h], p);
}

// outAcc[dst,h,:] += xl[src,h,:] * p/(den+eps)
__global__ void k_agg(const float* xl, const float* pvals, const float* den, const int* ei,
                      float* outAcc) {
    int e = blockIdx.x;
    int t = threadIdx.x;          // 512 threads = all (head,channel)
    int h = t >> 7;
    int s = (e < Ee) ? ei[e] : e - Ee;
    int d = (e < Ee) ? ei[Ee + e] : e - Ee;
    float coef = pvals[(size_t)e * 4 + h] / (den[(size_t)d * 4 + h] + 1e-16f);
    atomicAdd(&outAcc[(size_t)d * HC + t], xl[(size_t)s * HC + t] * coef);
}

// head mean + bias -> layernorm -> relu + residual ; one wave per node
__global__ void k_node_post(const float* outAcc, const float* bias, const float* g, const float* b,
                            float* hF32, _Float16* hF16) {
    int node = blockIdx.x * 4 + (threadIdx.x >> 5);
    int lane = threadIdx.x & 31;
    float v[4], hin[4];
#pragma unroll
    for (int j = 0; j < 4; j++) {
        int c = lane + j * 32;
        float s = 0.0f;
#pragma unroll
        for (int hh = 0; hh < 4; hh++) s += outAcc[(size_t)node * HC + hh * 128 + c];
        v[j] = 0.25f * s + bias[c];
        hin[j] = hF32[(size_t)node * Hh + c];
    }
    float sum = v[0] + v[1] + v[2] + v[3];
    sum = red32(sum);
    float mean = sum * (1.0f / 128.0f);
    float var = 0.0f;
#pragma unroll
    for (int j = 0; j < 4; j++) { float d = v[j] - mean; var += d * d; }
    var = red32(var) * (1.0f / 128.0f);
    float inv = rsqrtf(var + 1e-5f);
#pragma unroll
    for (int j = 0; j < 4; j++) {
        int c = lane + j * 32;
        float y = (v[j] - mean) * inv * g[c] + b[c];
        y = fmaxf(y, 0.0f) + hin[j];
        hF32[(size_t)node * Hh + c] = y;
        hF16[(size_t)node * Hh + c] = (_Float16)y;
    }
}

// final layernorm -> d_out
__global__ void k_final_ln(const float* hF32, const float* g, const float* b, float* out) {
    int node = blockIdx.x * 4 + (threadIdx.x >> 5);
    int lane = threadIdx.x & 31;
    float v[4];
#pragma unroll
    for (int j = 0; j < 4; j++) v[j] = hF32[(size_t)node * Hh + lane + j * 32];
    float sum = red32(v[0] + v[1] + v[2] + v[3]);
    float mean = sum * (1.0f / 128.0f);
    float var = 0.0f;
#pragma unroll
    for (int j = 0; j < 4; j++) { float d = v[j] - mean; var += d * d; }
    var = red32(var) * (1.0f / 128.0f);
    float inv = rsqrtf(var + 1e-5f);
#pragma unroll
    for (int j = 0; j < 4; j++) {
        int c = lane + j * 32;
        out[(size_t)node * Hh + c] = (v[j] - mean) * inv * g[c] + b[c];
    }
}

// ---------- launch ----------

extern "C" void kernel_launch(void* const* d_in, const int* in_sizes, int n_in,
                              void* d_out, int out_size, void* d_ws, size_t ws_size,
                              hipStream_t stream) {
    const int*   z    = (const int*)d_in[0];
    const float* pos  = (const float*)d_in[1];
    const int*   ei   = (const int*)d_in[2];
    const float* emb  = (const float*)d_in[3];
    const float* Wd   = (const float*)d_in[4];
    const float* bd   = (const float*)d_in[5];
    const float* Wl   = (const float*)d_in[6];
    const float* bl   = (const float*)d_in[7];
    const float* Wr   = (const float*)d_in[8];
    const float* br   = (const float*)d_in[9];
    const float* We   = (const float*)d_in[10];
    const float* att  = (const float*)d_in[11];
    const float* bias = (const float*)d_in[12];
    const float* ln_g = (const float*)d_in[13];
    const float* ln_b = (const float*)d_in[14];
    const float* fn_g = (const float*)d_in[15];
    const float* fn_b = (const float*)d_in[16];

    size_t off = 0;
    char* base = (char*)d_ws;
    auto alloc = [&](size_t bytes) -> char* {
        char* p = base + off;
        off = (off + bytes + 255) & ~(size_t)255;
        return p;
    };
    float*     hF32    = (float*)alloc((size_t)Nn * Hh * 4);
    _Float16*  hF16    = (_Float16*)alloc((size_t)Nn * Hh * 2);
    _Float16*  ea2     = (_Float16*)alloc((size_t)ET * Hh * 2);
    float*     loopSum = (float*)alloc((size_t)Nn * Hh * 4);
    float*     cnt     = (float*)alloc((size_t)Nn * 4);
    _Float16*  WlT     = (_Float16*)alloc((size_t)HC * Hh * 2);
    _Float16*  WrT     = (_Float16*)alloc((size_t)HC * Hh * 2);
    _Float16*  WeT     = (_Float16*)alloc((size_t)HC * Hh * 2);
    float*     xl      = (float*)alloc((size_t)Nn * HC * 4);
    float*     xr      = (float*)alloc((size_t)Nn * HC * 4);
    float*     logits  = (float*)alloc((size_t)ET * 4 * 4);
    float*     pvals   = (float*)alloc((size_t)ET * 4 * 4);
    unsigned*  amax    = (unsigned*)alloc((size_t)Nn * 4 * 4);
    float*     den     = (float*)alloc((size_t)Nn * 4 * 4);
    float*     outAcc  = (float*)alloc((size_t)Nn * HC * 4);

    // node embedding
    k_embed<<<(Nn * Hh + 255) / 256, 256, 0, stream>>>(z, emb, hF32, hF16);

    // edge features + self-loop means
    hipMemsetAsync(loopSum, 0, (size_t)Nn * Hh * 4, stream);
    hipMemsetAsync(cnt, 0, (size_t)Nn * 4, stream);
    k_edge_feat<<<Ee, 128, 0, stream>>>(pos, ei, Wd, bd, ea2, loopSum, cnt);
    k_loop_div<<<(Nn * Hh + 255) / 256, 256, 0, stream>>>(loopSum, cnt, ea2);

    for (int l = 0; l < LL; l++) {
        const float* Wl_l = Wl + (size_t)l * Hh * HC;
        const float* Wr_l = Wr + (size_t)l * Hh * HC;
        const float* We_l = We + (size_t)l * Hh * HC;
        const float* bl_l = bl + (size_t)l * HC;
        const float* br_l = br + (size_t)l * HC;
        const float* att_l  = att + (size_t)l * 4 * 128;
        const float* bias_l = bias + (size_t)l * 128;
        const float* g_l = ln_g + (size_t)l * Hh;
        const float* b_l = ln_b + (size_t)l * Hh;

        hipMemsetAsync(amax, 0, (size_t)Nn * 4 * 4, stream);   // enc space: 0 < enc(any real)
        hipMemsetAsync(den, 0, (size_t)Nn * 4 * 4, stream);
        hipMemsetAsync(outAcc, 0, (size_t)Nn * HC * 4, stream);

        k_wt<<<(HC * Hh + 255) / 256, 256, 0, stream>>>(Wl_l, WlT);
        k_wt<<<(HC * Hh + 255) / 256, 256, 0, stream>>>(Wr_l, WrT);
        k_wt<<<(HC * Hh + 255) / 256, 256, 0, stream>>>(We_l, WeT);

        dim3 gproj(Nn / 16, HC / 64);   // 625 x 8, 4 waves/block -> one 16x16 tile per wave
        k_gemm_proj<<<gproj, 128, 0, stream>>>(hF16, WlT, bl_l, xl);
        k_gemm_proj<<<gproj, 128, 0, stream>>>(hF16, WrT, br_l, xr);

        k_edge_logits<<<ET / (16 * TILES_PER_BLK), 128, HC * Hh * sizeof(_Float16), stream>>>(
            ea2, WeT, xl, xr, att_l, ei, logits, amax);
        k_softmax<<<(ET * 4 + 255) / 256, 256, 0, stream>>>(logits, amax, pvals, den, ei);
        k_agg<<<ET, 512, 0, stream>>>(xl, pvals, den, ei, outAcc);
        k_node_post<<<Nn / 4, 128, 0, stream>>>(outAcc, bias_l, g_l, b_l, hF32, hF16);
    }

    k_final_ln<<<Nn / 4, 128, 0, stream>>>(hF32, fn_g, fn_b, (float*)d_out);
}
